// CrossAttention_67216238182948
// MI455X (gfx1250) — matrix operand order repaired and checked by reading, not compiled, not run on hardware
//
#include <hip/hip_runtime.h>
#include <hip/hip_bf16.h>

typedef __attribute__((ext_vector_type(16))) _Float16 v16h;
typedef __attribute__((ext_vector_type(8)))  _Float16 v8h;
typedef __attribute__((ext_vector_type(8)))  float    v8f;

union V16 { v16h v; v8h h[2]; };

#define N_TOK   1024
#define CQ      256
#define CC      512
#define NHEAD   8
#define HDIM    32

// Use CDNA5 async global->LDS DMA (ASYNCcnt path) for GEMM A-tile staging.
// Set to 0 to fall back to the proven global_load_b128 + ds_store_b128 path.
#define GEMM_ASYNC_COPY 1

// generic shared-memory pointer -> LDS byte offset (low 32 bits of aperture addr)
__device__ __forceinline__ unsigned lds_off(const void* p) {
    return (unsigned)(unsigned long long)p;
}

__device__ __forceinline__ void async_copy_b128(void* lds_dst, const _Float16* gsrc) {
#if GEMM_ASYNC_COPY
    unsigned d = lds_off(lds_dst);
    asm volatile("global_load_async_to_lds_b128 %0, %1, off"
                 :: "v"(d), "v"(gsrc) : "memory");
#else
    *(v8h*)lds_dst = *(const v8h*)gsrc;
#endif
}

__device__ __forceinline__ void wait_async_copies() {
#if GEMM_ASYNC_COPY
    asm volatile("s_wait_asynccnt 0x0" ::: "memory");
#endif
}

// ---------------------------------------------------------------------------
// Kernel 1: adaptive 2x2 avg-pool of context + LayerNorm over 512 channels,
// emit f16 [B][N][512]
// ---------------------------------------------------------------------------
__global__ __launch_bounds__(256) void pool_ln_ctx_kernel(
    const float* __restrict__ ctx, const float* __restrict__ g,
    const float* __restrict__ be, _Float16* __restrict__ out)
{
    __shared__ float rs_[256], rq_[256];
    int bn = blockIdx.x;            // b*1024 + n
    int b  = bn >> 10, n = bn & 1023;
    int h  = n >> 5,   w = n & 31;
    int t  = threadIdx.x;
    float p[2]; float s = 0.f, sq = 0.f;
    #pragma unroll
    for (int i = 0; i < 2; ++i) {
        int c = t + i * 256;
        size_t base = (((size_t)(b * CC + c) * 64 + h * 2) * 64 + w * 2);
        float v = 0.25f * (ctx[base] + ctx[base + 1] + ctx[base + 64] + ctx[base + 65]);
        p[i] = v; s += v; sq += v * v;
    }
    rs_[t] = s; rq_[t] = sq; __syncthreads();
    for (int o = 128; o > 0; o >>= 1) {
        if (t < o) { rs_[t] += rs_[t + o]; rq_[t] += rq_[t + o]; }
        __syncthreads();
    }
    float mean = rs_[0] * (1.f / 512.f);
    float var  = rq_[0] * (1.f / 512.f) - mean * mean;
    float rinv = rsqrtf(var + 1e-5f);
    #pragma unroll
    for (int i = 0; i < 2; ++i) {
        int c = t + i * 256;
        out[(size_t)bn * CC + c] = (_Float16)((p[i] - mean) * rinv * g[c] + be[c]);
    }
}

// ---------------------------------------------------------------------------
// Kernel 2: LayerNorm of query over 256 channels -> f16 [B][N][256]
// ---------------------------------------------------------------------------
__global__ __launch_bounds__(256) void ln_query_kernel(
    const float* __restrict__ q, const float* __restrict__ g,
    const float* __restrict__ be, _Float16* __restrict__ out)
{
    __shared__ float rs_[256], rq_[256];
    int bn = blockIdx.x;
    int b  = bn >> 10, n = bn & 1023;
    int t  = threadIdx.x;
    float v = q[((size_t)b * CQ + t) * N_TOK + n];
    rs_[t] = v; rq_[t] = v * v; __syncthreads();
    for (int o = 128; o > 0; o >>= 1) {
        if (t < o) { rs_[t] += rs_[t + o]; rq_[t] += rq_[t + o]; }
        __syncthreads();
    }
    float mean = rs_[0] * (1.f / 256.f);
    float var  = rq_[0] * (1.f / 256.f) - mean * mean;
    float rinv = rsqrtf(var + 1e-5f);
    out[(size_t)bn * CQ + t] = (_Float16)((v - mean) * rinv * g[t] + be[t]);
}

// ---------------------------------------------------------------------------
// Kernel 3: f32 -> f16 weight conversion WITH transpose:
//   src W[k][n]  (K x Nmat, row-major f32)
//   dst WT[n][k] (Nmat x K, row-major f16)
// ---------------------------------------------------------------------------
__global__ void cvt_transpose_f16_kernel(const float* __restrict__ s,
                                         _Float16* __restrict__ d,
                                         int K, int Nmat)
{
    int i = blockIdx.x * 256 + threadIdx.x;
    int total = K * Nmat;
    if (i < total) {
        int k = i / Nmat, n = i % Nmat;
        d[(size_t)n * K + k] = (_Float16)s[i];
    }
}

// ---------------------------------------------------------------------------
// Kernel 4: batched WMMA GEMM: Out[b] = A[b] (MxK f16) @ W (KxN) + bias,
// weights pre-transposed WT[n][k]. Block tile 64x64, 4 waves of 2x2
// wmma 16x16x32 tiles. A-tile staged via async global->LDS DMA; B fragments
// loaded directly from global (L2-resident weights, contiguous 32B/lane).
// transOut=1 writes Out as [b][n][m] (used for V).
// ---------------------------------------------------------------------------
__global__ __launch_bounds__(128) void gemm_qkv_kernel(
    const _Float16* __restrict__ A, const _Float16* __restrict__ WT,
    const float* __restrict__ bias, _Float16* __restrict__ Out,
    int M, int Nmat, int K, int transOut)
{
    __shared__ _Float16 As[64 * 32];     // [row][k]
    int b   = blockIdx.z;
    int m0  = blockIdx.y * 64, n0 = blockIdx.x * 64;
    int tid = threadIdx.x;
    int wave = tid >> 5, lane = tid & 31;
    int wm = (wave >> 1) * 32, wn = (wave & 1) * 32;
    int l15 = lane & 15, hi = lane >> 4;
    const _Float16* Ab = A + (size_t)b * M * K;

    v8f zero = {};
    v8f acc[2][2];
    #pragma unroll
    for (int i = 0; i < 2; ++i)
        #pragma unroll
        for (int j = 0; j < 2; ++j) acc[i][j] = zero;

    int rA = tid >> 1, kA = (tid & 1) * 16;    // this thread's 16-half slice
    for (int k0 = 0; k0 < K; k0 += 32) {
        // stage A tile 64x32 via async DMA (2 x b128 per thread)
        {
            const _Float16* src = &Ab[(size_t)(m0 + rA) * K + k0 + kA];
            async_copy_b128(&As[rA * 32 + kA],     src);
            async_copy_b128(&As[rA * 32 + kA + 8], src + 8);
        }
        if (k0 + 32 < K) {  // prefetch next K-chunk (global_prefetch_b8)
            __builtin_prefetch(&Ab[(size_t)(m0 + rA) * K + k0 + 32], 0, 1);
            __builtin_prefetch(&WT[(size_t)(n0 + (tid & 63)) * K + k0 + 32], 0, 1);
        }
        // B fragments straight from global: overlaps with the A-tile DMA
        V16 bf[2];
        #pragma unroll
        for (int j = 0; j < 2; ++j) {
            const _Float16* wrow = WT + (size_t)(n0 + wn + j * 16 + l15) * K + k0 + hi * 16;
            bf[j].h[0] = *(const v8h*)&wrow[0];
            bf[j].h[1] = *(const v8h*)&wrow[8];
        }
        wait_async_copies();
        __syncthreads();

        V16 af[2];
        #pragma unroll
        for (int i = 0; i < 2; ++i) {
            int row = wm + i * 16 + l15;
            af[i].h[0] = *(const v8h*)&As[row * 32 + hi * 16];
            af[i].h[1] = *(const v8h*)&As[row * 32 + hi * 16 + 8];
        }
        #pragma unroll
        for (int i = 0; i < 2; ++i)
            #pragma unroll
            for (int j = 0; j < 2; ++j)
                acc[i][j] = __builtin_amdgcn_wmma_f32_16x16x32_f16(
                    false, af[i].v, false, bf[j].v, (short)0, acc[i][j], false, false);
        __syncthreads();
    }

    // epilogue: + bias, store f16 (optionally transposed)
    #pragma unroll
    for (int i = 0; i < 2; ++i)
        #pragma unroll
        for (int j = 0; j < 2; ++j) {
            int nbase = n0 + wn + j * 16 + l15;
            float bv = bias[nbase];
            #pragma unroll
            for (int r = 0; r < 8; ++r) {
                int mrow = m0 + wm + i * 16 + r + 8 * hi;
                float v = acc[i][j][r] + bv;
                size_t o = transOut
                    ? ((size_t)b * M * Nmat + (size_t)nbase * M + mrow)
                    : ((size_t)b * M * Nmat + (size_t)mrow * Nmat + nbase);
                Out[o] = (_Float16)v;
            }
        }
}

// ---------------------------------------------------------------------------
// Kernel 5: flash attention. One wave = 16 query rows of one (b,head).
// Q,K: [B][N][256] f16 ; V: [B][256][N] f16 (transposed) ; out AO [B][N][256]
// ---------------------------------------------------------------------------
__global__ __launch_bounds__(128) void attn_kernel(
    const _Float16* __restrict__ Qh, const _Float16* __restrict__ Kh,
    const _Float16* __restrict__ Vt, _Float16* __restrict__ AO)
{
    __shared__ _Float16 Pst[4][16 * 32];  // per-wave P staging (f16, C-layout -> A-layout)
    int bh = blockIdx.x; int b = bh >> 3, h = bh & 7;
    int wave = threadIdx.x >> 5, lane = threadIdx.x & 31;
    int l15 = lane & 15, hi = lane >> 4;
    int q0 = (blockIdx.y * 4 + wave) * 16;
    const _Float16* Qb = Qh + (size_t)b * N_TOK * CQ + h * HDIM;
    const _Float16* Kb = Kh + (size_t)b * N_TOK * CQ + h * HDIM;
    const _Float16* Vb = Vt + ((size_t)b * CQ + h * HDIM) * N_TOK;
    _Float16* P = Pst[wave];

    V16 qf;
    {
        const _Float16* qrow = Qb + (size_t)(q0 + l15) * CQ;
        qf.h[0] = *(const v8h*)&qrow[hi * 16];
        qf.h[1] = *(const v8h*)&qrow[hi * 16 + 8];
    }
    v8f zero = {};
    v8f o0 = zero, o1 = zero;
    float m[8], l[8];
    #pragma unroll
    for (int r = 0; r < 8; ++r) { m[r] = -1e30f; l[r] = 0.f; }
    const float rs = 0.17677669529663689f;   // 1/sqrt(32)

    for (int kc = 0; kc < N_TOK; kc += 32) {
        // S tiles: K^T fragments are contiguous 32B per lane
        V16 kf0, kf1;
        {
            const _Float16* k0p = Kb + (size_t)(kc + l15) * CQ + hi * 16;
            kf0.h[0] = *(const v8h*)&k0p[0];
            kf0.h[1] = *(const v8h*)&k0p[8];
            const _Float16* k1p = Kb + (size_t)(kc + 16 + l15) * CQ + hi * 16;
            kf1.h[0] = *(const v8h*)&k1p[0];
            kf1.h[1] = *(const v8h*)&k1p[8];
        }
        v8f s0 = __builtin_amdgcn_wmma_f32_16x16x32_f16(false, qf.v, false, kf0.v, (short)0, zero, false, false);
        v8f s1 = __builtin_amdgcn_wmma_f32_16x16x32_f16(false, qf.v, false, kf1.v, (short)0, zero, false, false);

        // online softmax per row (row r+8*hi lives in this half-wave)
        #pragma unroll
        for (int r = 0; r < 8; ++r) {
            float a0 = s0[r] * rs, a1 = s1[r] * rs;
            float mx = fmaxf(a0, a1);
            #pragma unroll
            for (int d = 8; d >= 1; d >>= 1) mx = fmaxf(mx, __shfl_xor(mx, d, 16));
            float mnew  = fmaxf(m[r], mx);
            float alpha = __expf(m[r] - mnew);
            float p0 = __expf(a0 - mnew), p1 = __expf(a1 - mnew);
            float sum = p0 + p1;
            #pragma unroll
            for (int d = 8; d >= 1; d >>= 1) sum += __shfl_xor(sum, d, 16);
            l[r] = l[r] * alpha + sum;
            m[r] = mnew;
            o0[r] *= alpha; o1[r] *= alpha;
            P[(r + 8 * hi) * 32 + l15]      = (_Float16)p0;
            P[(r + 8 * hi) * 32 + 16 + l15] = (_Float16)p1;
        }
        // re-read P in A-fragment layout: two ds_load_b128 per lane
        // (same-wave DS ops are in-order; buffer is wave-private, no barrier)
        V16 pf;
        {
            const _Float16* prow = &P[l15 * 32];
            pf.h[0] = *(const v8h*)&prow[hi * 16];
            pf.h[1] = *(const v8h*)&prow[hi * 16 + 8];
        }
        // V fragments from transposed V: contiguous 32B per lane
        V16 vf0, vf1;
        {
            const _Float16* v0p = Vb + (size_t)l15 * N_TOK + kc + hi * 16;
            vf0.h[0] = *(const v8h*)&v0p[0];
            vf0.h[1] = *(const v8h*)&v0p[8];
            const _Float16* v1p = Vb + (size_t)(16 + l15) * N_TOK + kc + hi * 16;
            vf1.h[0] = *(const v8h*)&v1p[0];
            vf1.h[1] = *(const v8h*)&v1p[8];
        }
        o0 = __builtin_amdgcn_wmma_f32_16x16x32_f16(false, pf.v, false, vf0.v, (short)0, o0, false, false);
        o1 = __builtin_amdgcn_wmma_f32_16x16x32_f16(false, pf.v, false, vf1.v, (short)0, o1, false, false);
    }

    _Float16* Ob = AO + (size_t)b * N_TOK * CQ + h * HDIM;
    #pragma unroll
    for (int r = 0; r < 8; ++r) {
        float inv = 1.f / l[r];
        int row = q0 + r + 8 * hi;
        Ob[(size_t)row * CQ + l15]      = (_Float16)(o0[r] * inv);
        Ob[(size_t)row * CQ + 16 + l15] = (_Float16)(o1[r] * inv);
    }
}

// ---------------------------------------------------------------------------
// Kernel 6: output projection + bias + residual, transposed f32 store
// out[b][c][n] = (AO[b] @ Wo)[n_tok][c] + bo[c] + query[b][c][n_tok]
// Weights pre-transposed: WT[n][k].
// ---------------------------------------------------------------------------
__global__ __launch_bounds__(128) void out_proj_kernel(
    const _Float16* __restrict__ A, const _Float16* __restrict__ WT,
    const float* __restrict__ bias, const float* __restrict__ query,
    float* __restrict__ out)
{
    __shared__ _Float16 As[64 * 32];
    const int M = N_TOK, Nmat = CQ, K = CQ;
    int b   = blockIdx.z;
    int m0  = blockIdx.y * 64, n0 = blockIdx.x * 64;
    int tid = threadIdx.x;
    int wave = tid >> 5, lane = tid & 31;
    int wm = (wave >> 1) * 32, wn = (wave & 1) * 32;
    int l15 = lane & 15, hi = lane >> 4;
    const _Float16* Ab = A + (size_t)b * M * K;

    v8f zero = {};
    v8f acc[2][2];
    #pragma unroll
    for (int i = 0; i < 2; ++i)
        #pragma unroll
        for (int j = 0; j < 2; ++j) acc[i][j] = zero;

    int rA = tid >> 1, kA = (tid & 1) * 16;
    for (int k0 = 0; k0 < K; k0 += 32) {
        {
            const _Float16* src = &Ab[(size_t)(m0 + rA) * K + k0 + kA];
            async_copy_b128(&As[rA * 32 + kA],     src);
            async_copy_b128(&As[rA * 32 + kA + 8], src + 8);
        }
        if (k0 + 32 < K) {
            __builtin_prefetch(&Ab[(size_t)(m0 + rA) * K + k0 + 32], 0, 1);
        }
        V16 bf[2];
        #pragma unroll
        for (int j = 0; j < 2; ++j) {
            const _Float16* wrow = WT + (size_t)(n0 + wn + j * 16 + l15) * K + k0 + hi * 16;
            bf[j].h[0] = *(const v8h*)&wrow[0];
            bf[j].h[1] = *(const v8h*)&wrow[8];
        }
        wait_async_copies();
        __syncthreads();

        V16 af[2];
        #pragma unroll
        for (int i = 0; i < 2; ++i) {
            int row = wm + i * 16 + l15;
            af[i].h[0] = *(const v8h*)&As[row * 32 + hi * 16];
            af[i].h[1] = *(const v8h*)&As[row * 32 + hi * 16 + 8];
        }
        #pragma unroll
        for (int i = 0; i < 2; ++i)
            #pragma unroll
            for (int j = 0; j < 2; ++j)
                acc[i][j] = __builtin_amdgcn_wmma_f32_16x16x32_f16(
                    false, af[i].v, false, bf[j].v, (short)0, acc[i][j], false, false);
        __syncthreads();
    }

    #pragma unroll
    for (int i = 0; i < 2; ++i)
        #pragma unroll
        for (int j = 0; j < 2; ++j) {
            int nbase = n0 + wn + j * 16 + l15;
            float bv = bias[nbase];
            #pragma unroll
            for (int r = 0; r < 8; ++r) {
                int mrow = m0 + wm + i * 16 + r + 8 * hi;
                size_t oi = ((size_t)b * CQ + nbase) * N_TOK + mrow;
                out[oi] = acc[i][j][r] + bv + query[oi];
            }
        }
}

// ---------------------------------------------------------------------------
extern "C" void kernel_launch(void* const* d_in, const int* in_sizes, int n_in,
                              void* d_out, int out_size, void* d_ws, size_t ws_size,
                              hipStream_t stream) {
    const float* query   = (const float*)d_in[0];
    const float* context = (const float*)d_in[1];
    const float* Wq      = (const float*)d_in[2];
    const float* bq      = (const float*)d_in[3];
    const float* Wk      = (const float*)d_in[4];
    const float* bk      = (const float*)d_in[5];
    const float* Wv      = (const float*)d_in[6];
    const float* bv      = (const float*)d_in[7];
    const float* Wo      = (const float*)d_in[8];
    const float* bo      = (const float*)d_in[9];
    const float* gq      = (const float*)d_in[10];
    const float* betq    = (const float*)d_in[11];
    const float* gc      = (const float*)d_in[12];
    const float* betc    = (const float*)d_in[13];
    float* out = (float*)d_out;

    const int B = 8;
    char* ws = (char*)d_ws;
    size_t off = 0;
    auto bump = [&](size_t bytes) {
        size_t p = off; off += (bytes + 255) & ~(size_t)255; return (void*)(ws + p);
    };
    _Float16* qnh = (_Float16*)bump((size_t)B * N_TOK * CQ * 2);
    _Float16* cnh = (_Float16*)bump((size_t)B * N_TOK * CC * 2);
    _Float16* wqh = (_Float16*)bump((size_t)CQ * CQ * 2);   // transposed [256][256]
    _Float16* wkh = (_Float16*)bump((size_t)CC * CQ * 2);   // transposed [256][512]
    _Float16* wvh = (_Float16*)bump((size_t)CC * CQ * 2);   // transposed [256][512]
    _Float16* woh = (_Float16*)bump((size_t)CQ * CQ * 2);   // transposed [256][256]
    _Float16* Qf  = (_Float16*)bump((size_t)B * N_TOK * CQ * 2);
    _Float16* Kf  = (_Float16*)bump((size_t)B * N_TOK * CQ * 2);
    _Float16* Vtf = (_Float16*)bump((size_t)B * N_TOK * CQ * 2);  // transposed
    _Float16* AO  = (_Float16*)bump((size_t)B * N_TOK * CQ * 2);

    // 1) normalizations
    ln_query_kernel<<<dim3(B * N_TOK), 256, 0, stream>>>(query, gq, betq, qnh);
    pool_ln_ctx_kernel<<<dim3(B * N_TOK), 256, 0, stream>>>(context, gc, betc, cnh);

    // 2) weight conversion + transpose (one-time)
    cvt_transpose_f16_kernel<<<dim3((CQ * CQ + 255) / 256), 256, 0, stream>>>(Wq, wqh, CQ, CQ);
    cvt_transpose_f16_kernel<<<dim3((CC * CQ + 255) / 256), 256, 0, stream>>>(Wk, wkh, CC, CQ);
    cvt_transpose_f16_kernel<<<dim3((CC * CQ + 255) / 256), 256, 0, stream>>>(Wv, wvh, CC, CQ);
    cvt_transpose_f16_kernel<<<dim3((CQ * CQ + 255) / 256), 256, 0, stream>>>(Wo, woh, CQ, CQ);

    // 3) Q/K/V projections (WMMA)
    dim3 ggrid(CQ / 64, N_TOK / 64, B);   // (4,16,8)
    gemm_qkv_kernel<<<ggrid, 128, 0, stream>>>(qnh, wqh, bq, Qf,  N_TOK, CQ, CQ, 0);
    gemm_qkv_kernel<<<ggrid, 128, 0, stream>>>(cnh, wkh, bk, Kf,  N_TOK, CQ, CC, 0);
    gemm_qkv_kernel<<<ggrid, 128, 0, stream>>>(cnh, wvh, bv, Vtf, N_TOK, CQ, CC, 1);

    // 4) attention (WMMA flash)
    attn_kernel<<<dim3(B * NHEAD, N_TOK / 64), 128, 0, stream>>>(Qf, Kf, Vtf, AO);

    // 5) output projection + residual + transpose (WMMA)
    out_proj_kernel<<<ggrid, 128, 0, stream>>>(AO, woh, bo, query, out);

    (void)in_sizes; (void)n_in; (void)out_size; (void)ws_size;
}
